// PartialConvolution_34445637714646
// MI455X (gfx1250) — matrix operand order repaired and checked by reading, not compiled, not run on hardware
//
#include <hip/hip_runtime.h>
#include <hip/hip_bf16.h>

// ---------------------------------------------------------------------------
// PartialConvolution on MI455X (gfx1250, wave32, WMMA).
//
// out[p,d] = norm[p] * sum_{k,c} sel[p,k] * x[nb(p,k),c] * w[c,k,d]
//  -> GEMM  M=25600 (pixels), K=2304 (9 taps x 256 ch), N=256
// Compute-bound (~500 FLOP/byte): K-loop on V_WMMA_F32_16X16X32_F16,
// f16 inputs / f32 accumulation.  Working set (x_f16 13MB + packed w 1.2MB)
// lives in the 192MB L2; B fragments are reused by all M-tiles.
// Each wave: 2 A fragments (32 M rows) x 4 N-tiles -> 8 WMMAs per K-step.
//
// Border taps: reference guarantees sel[p,t]==0 for out-of-image neighbors
// (zero-padded seg patch), so A = x*sel is zero there automatically; we only
// clamp the load address into the image for memory safety -> branch-free
// inner loop (no EXEC masking around the A loads).
// ---------------------------------------------------------------------------

#define BATCH 4
#define H_   80
#define W_   80
#define CH_  256
#define NC_  22
#define DIM_ 256
#define NPIX (BATCH * H_ * W_)          // 25600
#define KTOT (9 * CH_)                  // 2304

typedef __attribute__((ext_vector_type(16))) _Float16 v16h;
typedef __attribute__((ext_vector_type(8)))  _Float16 v8h;
typedef __attribute__((ext_vector_type(8)))  float    v8f;

// -------------------- workspace layout (bytes) ------------------------------
#define X16_BYTES  ((size_t)NPIX * CH_ * 2)            // 13,107,200
#define WF_BYTES   ((size_t)KTOT * DIM_ * 2)           //  1,179,648
#define SEL_BYTES  ((size_t)NPIX * 9 * 4)              //    921,600
#define NRM_BYTES  ((size_t)NPIX * 4)                  //    102,400

// -------------------- kernel 1: x f32 -> f16 --------------------------------
__global__ void k_cvt_x(const float* __restrict__ x, _Float16* __restrict__ x16,
                        int n) {
    int i = (blockIdx.x * blockDim.x + threadIdx.x) * 4;
    if (i + 3 < n) {
        float4 v = *(const float4*)(x + i);
        x16[i + 0] = (_Float16)v.x;
        x16[i + 1] = (_Float16)v.y;
        x16[i + 2] = (_Float16)v.z;
        x16[i + 3] = (_Float16)v.w;
    }
}

// -------------------- kernel 2: repack conv_w into B-fragment order ---------
// Fragment f = (t*8 + cb)*16 + ntile, 512 f16 each.  Within a fragment,
// element index = lane*16 + e, where (wave32 16-bit B 32x16 layout):
//   K = 16*(e>=8) + 8*(lane>=16) + (e&7)   (channel c = cb*32 + K),   N = lane & 15.
// Each GEMM lane then reads its whole v16h B fragment contiguously (32B).
__global__ void k_pack_w(const float* __restrict__ cw, _Float16* __restrict__ wf) {
    int e = blockIdx.x * blockDim.x + threadIdx.x;     // 0 .. 589823
    if (e >= KTOT * DIM_) return;
    int frag = e >> 9;
    int r    = e & 511;
    int lane = r >> 4;
    int el   = r & 15;
    int nt = frag & 15;
    int cb = (frag >> 4) & 7;
    int t  = frag >> 7;
    int K = ((el >> 3) << 4) + ((lane >> 4) << 3) + (el & 7);
    int c = cb * 32 + K;
    int d = nt * 16 + (lane & 15);
    // conv_w shape (CH,3,3,DIM): w[c,t,d] = cw[(c*9 + t)*DIM + d]
    wf[e] = (_Float16)cw[(size_t)(c * 9 + t) * DIM_ + d];
}

// -------------------- kernel 3: sel + norm per pixel ------------------------
__global__ void k_selnorm(const float* __restrict__ seg,
                          float* __restrict__ sel, float* __restrict__ nrm) {
    int p = blockIdx.x * blockDim.x + threadIdx.x;
    if (p >= NPIX) return;
    int b = p / (H_ * W_);
    int q = p % (H_ * W_);
    int h = q / W_;
    int w = q % W_;
    const float* sc = seg + (size_t)p * NC_;
    float mv = sc[0];
    #pragma unroll
    for (int c = 1; c < NC_; ++c) mv = fmaxf(mv, sc[c]);
    int cnt = 0;
    int t = 0;
    for (int dy = -1; dy <= 1; ++dy) {
        for (int dx = -1; dx <= 1; ++dx, ++t) {
            int hn = h + dy;
            int wn = w + dx;
            float s = 0.0f;
            if ((unsigned)hn < (unsigned)H_ && (unsigned)wn < (unsigned)W_) {
                const float* sn = seg + (size_t)(b * H_ * W_ + hn * W_ + wn) * NC_;
                for (int c = 0; c < NC_; ++c)
                    s += (sc[c] == mv) ? sn[c] : 0.0f;   // exact tie-summing
            }
            sel[(size_t)p * 9 + t] = s;                  // 0 for padded taps
            if (s != 0.0f) ++cnt;
        }
    }
    nrm[p] = (cnt > 0) ? 9.0f / (float)cnt : 0.0f;
}

// -------------------- kernel 4: WMMA GEMM -----------------------------------
// block = 128 threads = 4 waves.  Block owns one 32-pixel M-tile; each wave
// owns 4 N-tiles (64 output channels) for BOTH 16-row halves.
// K-loop: 9 taps (rolled, cheap dy/dx counters) x 8 channel-blocks (unrolled),
// 8x v_wmma_f32_16x16x32_f16 per step (576 WMMAs per wave).
__global__ __launch_bounds__(128) void k_pconv_gemm(
        const _Float16* __restrict__ x16, const _Float16* __restrict__ wf,
        const float* __restrict__ sel, const float* __restrict__ nrm,
        float* __restrict__ out) {
    const int lane  = threadIdx.x & 31;
    const int wave  = threadIdx.x >> 5;       // 0..3 -> N group
    const int pbase = blockIdx.x * 32;        // 0..799 M-tiles of 32 pixels
    const int m  = lane & 15;                 // A row (both lane halves: M=0..15)
    const int kh = lane >> 4;                 // K-half: 0 -> K{0..7,16..23}, 1 -> K{8..15,24..31}
    const int nbase = wave * 4;               // first N-tile of this wave

    // two pixel rows per lane: A0 covers pixels pbase+0..15, A1 covers +16..31
    const int p0 = pbase + m;
    const int p1 = p0 + 16;
    const int b0 = p0 / (H_ * W_), q0 = p0 % (H_ * W_);
    const int h0 = q0 / W_,        w0 = q0 % W_;
    const int b1 = p1 / (H_ * W_), q1 = p1 % (H_ * W_);
    const int h1 = q1 / W_,        w1 = q1 % W_;
    const float* selp0 = sel + (size_t)p0 * 9;
    const float* selp1 = sel + (size_t)p1 * 9;

    v8f acc[2][4];
    #pragma unroll
    for (int g = 0; g < 2; ++g)
        #pragma unroll
        for (int n = 0; n < 4; ++n) acc[g][n] = {};

    int t = 0;
    for (int dy = -1; dy <= 1; ++dy) {
        for (int dx = -1; dx <= 1; ++dx, ++t) {
            // ---- per-tap setup (hoisted out of the K inner loop) ----
            // Clamp neighbor coords into the image: for out-of-image taps
            // sel==0 already zeroes the A fragment, clamp is only for safety.
            const int hn0 = min(max(h0 + dy, 0), H_ - 1);
            const int wn0 = min(max(w0 + dx, 0), W_ - 1);
            const int hn1 = min(max(h1 + dy, 0), H_ - 1);
            const int wn1 = min(max(w1 + dx, 0), W_ - 1);
            const _Float16* xr0 =
                x16 + ((size_t)(b0 * H_ * W_ + hn0 * W_ + wn0) << 8) + kh * 8;
            const _Float16* xr1 =
                x16 + ((size_t)(b1 * H_ * W_ + hn1 * W_ + wn1) << 8) + kh * 8;
            const _Float16 s0 = (_Float16)selp0[t];
            const _Float16 s1 = (_Float16)selp1[t];
            const _Float16* wbt =
                wf + (size_t)(t * 8 * 16 + nbase) * 512 + lane * 16;

            #pragma unroll
            for (int cb = 0; cb < 8; ++cb) {
                // ---- A fragments (ISA 16-bit A 16x32 layout), branch-free ----
                v8h a0lo = *(const v8h*)(xr0 + cb * 32);       // K = kh*8 + 0..7
                v8h a0hi = *(const v8h*)(xr0 + cb * 32 + 16);  // K = 16+kh*8+0..7
                v8h a1lo = *(const v8h*)(xr1 + cb * 32);
                v8h a1hi = *(const v8h*)(xr1 + cb * 32 + 16);
                v16h A0, A1;
                #pragma unroll
                for (int i = 0; i < 8; ++i) {                  // packed f16 muls
                    A0[i]     = a0lo[i] * s0;
                    A0[8 + i] = a0hi[i] * s0;
                    A1[i]     = a1lo[i] * s1;
                    A1[8 + i] = a1hi[i] * s1;
                }
                // ---- 4 B fragments, each feeds 2 WMMAs ----
                const _Float16* wb = wbt + (size_t)cb * 16 * 512;
                #pragma unroll
                for (int n = 0; n < 4; ++n) {
                    v16h Bf = *(const v16h*)(wb + n * 512);
                    acc[0][n] = __builtin_amdgcn_wmma_f32_16x16x32_f16(
                                    false, A0, false, Bf, (short)0, acc[0][n],
                                    false, false);
                    acc[1][n] = __builtin_amdgcn_wmma_f32_16x16x32_f16(
                                    false, A1, false, Bf, (short)0, acc[1][n],
                                    false, false);
                }
            }
        }
    }

    // ---- epilogue: C/D layout VGPR r -> M = r + 8*kh, N = lane&15 ----
    const int col = lane & 15;
    #pragma unroll
    for (int g = 0; g < 2; ++g) {
        const int rbase = pbase + g * 16 + kh * 8;
        #pragma unroll
        for (int r = 0; r < 8; ++r) {
            const float nv = nrm[rbase + r];
            float* orow = out + (size_t)(rbase + r) * DIM_ + nbase * 16 + col;
            orow[0]  = acc[g][0][r] * nv;
            orow[16] = acc[g][1][r] * nv;
            orow[32] = acc[g][2][r] * nv;
            orow[48] = acc[g][3][r] * nv;
        }
    }
}

// -------------------- launcher ----------------------------------------------
extern "C" void kernel_launch(void* const* d_in, const int* in_sizes, int n_in,
                              void* d_out, int out_size, void* d_ws, size_t ws_size,
                              hipStream_t stream) {
    const float* x   = (const float*)d_in[0];   // (4,80,80,256)
    const float* seg = (const float*)d_in[1];   // (4,80,80,22)
    const float* cw  = (const float*)d_in[2];   // (256,3,3,256)
    float* out = (float*)d_out;                 // (4,80,80,256)

    char* ws = (char*)d_ws;
    _Float16* x16 = (_Float16*)ws;
    _Float16* wf  = (_Float16*)(ws + X16_BYTES);
    float*    sel = (float*)(ws + X16_BYTES + WF_BYTES);
    float*    nrm = (float*)(ws + X16_BYTES + WF_BYTES + SEL_BYTES);

    const int nx = NPIX * CH_;                  // 6,553,600
    k_cvt_x<<<nx / (256 * 4), 256, 0, stream>>>(x, x16, nx);
    k_pack_w<<<(KTOT * DIM_) / 256, 256, 0, stream>>>(cw, wf);
    k_selnorm<<<(NPIX + 127) / 128, 128, 0, stream>>>(seg, sel, nrm);
    k_pconv_gemm<<<NPIX / 32, 128, 0, stream>>>(x16, wf, sel, nrm, out);
}
